// LinkDecoder_7816840479176
// MI455X (gfx1250) — matrix operand order repaired and checked by reading, not compiled, not run on hardware
//
#include <hip/hip_runtime.h>

typedef __attribute__((ext_vector_type(2)))  __bf16 v2bf;
typedef __attribute__((ext_vector_type(4)))  __bf16 v4bf;
typedef __attribute__((ext_vector_type(8)))  __bf16 v8bf;
typedef __attribute__((ext_vector_type(16))) __bf16 v16bf;
typedef __attribute__((ext_vector_type(8)))  float  v8f;

#define IN_DIM          128
#define HID_DIM         128
#define EDGES_PER_WAVE  16
#define WAVES_PER_BLOCK 8
#define EDGES_PER_BLOCK (EDGES_PER_WAVE * WAVES_PER_BLOCK)   // 128
#define TILES_PER_BLOCK 4                                    // 512 edges per block
#define THREADS         (WAVES_PER_BLOCK * 32)

// Padded strides (halves): 128 + 8 -> bank index becomes (row*4 + word) % 64,
// a stride-4 pattern => conflict-free b128 fragment loads. Pad keeps 16B alignment.
#define KSTRIDE 136

// LDS layout (dynamic shared):
//   sW1 : bf16 [128 cols][KSTRIDE]  column-major (sW1[n*KSTRIDE + k])  34816 B
//   sX  : bf16 [8 waves][16 edges][KSTRIDE]                            34816 B
//   sB1 : f32  [128]                                                     512 B
//   sW2 : f32  [128]                                                     512 B
#define SMEM_W1   0
#define SMEM_X    34816
#define SMEM_B1   (34816 * 2)
#define SMEM_W2   (34816 * 2 + 512)
#define SMEM_SIZE (34816 * 2 + 1024)

__global__ __launch_bounds__(THREADS)
void link_decoder_kernel(const float* __restrict__ h,
                         const long long* __restrict__ eidx,
                         const float* __restrict__ W1,
                         const float* __restrict__ b1,
                         const float* __restrict__ W2,
                         const float* __restrict__ b2,
                         float* __restrict__ out,
                         long long E)
{
    extern __shared__ char smem[];
    __bf16* sW1 = (__bf16*)(smem + SMEM_W1);
    __bf16* sX  = (__bf16*)(smem + SMEM_X);
    float*  sB1 = (float*)(smem + SMEM_B1);
    float*  sW2 = (float*)(smem + SMEM_W2);

    const int tid  = threadIdx.x;
    const int lane = tid & 31;
    const int wave = tid >> 5;

    // ---- Stage W1 once per block: transpose to column-major bf16, packed k-pairs ----
    // Thread handles (n, k, k+1): global reads coalesced across n; one ds_store_b32.
    for (int i = tid; i < (IN_DIM / 2) * HID_DIM; i += THREADS) {
        int kp = i >> 7;          // k-pair index 0..63
        int n  = i & 127;         // hidden column
        float a = W1[(2 * kp)     * HID_DIM + n];
        float bq = W1[(2 * kp + 1) * HID_DIM + n];
        v2bf p;
        p.x = (__bf16)a;
        p.y = (__bf16)bq;
        *(v2bf*)(sW1 + n * KSTRIDE + 2 * kp) = p;
    }
    if (tid < HID_DIM) {
        sB1[tid] = b1[tid];
        sW2[tid] = W2[tid];
    }
    __syncthreads();

    const int m  = lane & 15;     // fragment row / output column within tile
    const int hi = lane >> 4;     // half-wave select
    __bf16* xt = sX + wave * (EDGES_PER_WAVE * KSTRIDE);
    const __bf16* arow = xt + m * KSTRIDE;
    const float b2v = b2[0];

    const long long blockBase =
        (long long)blockIdx.x * (EDGES_PER_BLOCK * TILES_PER_BLOCK);

    for (int t = 0; t < TILES_PER_BLOCK; ++t) {
        const long long ebase =
            blockBase + (long long)t * EDGES_PER_BLOCK + (long long)wave * EDGES_PER_WAVE;
        if (ebase >= E) break;    // wave-uniform: EXEC stays all-ones

        // ---- Gather + fuse: x = h[u] * h[v] -> bf16 A-tile in LDS (per-wave) ----
        #pragma unroll 4
        for (int e = 0; e < EDGES_PER_WAVE; ++e) {
            long long edge = ebase + e;
            long long ec   = edge < E ? edge : (E - 1);  // clamp tail
            long long u = eidx[ec];
            long long v = eidx[E + ec];
            float4 hu = *(const float4*)(h + u * IN_DIM + lane * 4);
            float4 hv = *(const float4*)(h + v * IN_DIM + lane * 4);
            v4bf p;
            p.x = (__bf16)(hu.x * hv.x);
            p.y = (__bf16)(hu.y * hv.y);
            p.z = (__bf16)(hu.z * hv.z);
            p.w = (__bf16)(hu.w * hv.w);
            *(v4bf*)(xt + e * KSTRIDE + lane * 4) = p;
        }

        // ---- A fragments (ISA 16-bit 16x32 A layout) ----
        // lanes 0-15 : M = lane,    v0..3 = K 0..7,  v4..7 = K 16..23 (per k-step)
        // lanes 16-31: M = lane-16, v0..3 = K 8..15, v4..7 = K 24..31
        v16bf afrag[4];
        #pragma unroll
        for (int kt = 0; kt < 4; ++kt) {
            int kb = kt * 32 + hi * 8;
            ((v8bf*)&afrag[kt])[0] = *(const v8bf*)(arow + kb);
            ((v8bf*)&afrag[kt])[1] = *(const v8bf*)(arow + kb + 16);
        }

        // ---- GEMM1 (bf16 WMMA, f32 acc) fused with bias/ReLU/layer-2 dot ----
        float s[8];
        #pragma unroll
        for (int r = 0; r < 8; ++r) s[r] = 0.0f;

        #pragma unroll
        for (int nt = 0; nt < 8; ++nt) {
            const int n = nt * 16 + m;                  // this lane's output column
            const __bf16* bcol = sW1 + n * KSTRIDE;     // contiguous K, conflict-free
            v8f acc = {};
            #pragma unroll
            for (int kt = 0; kt < 4; ++kt) {
                // B fragment (32x16 bf16): lanes 0-15 K 0..7,16..23, lanes 16-31 K 8..15,24..31
                int kb = kt * 32 + hi * 8;
                v16bf bfrag;
                ((v8bf*)&bfrag)[0] = *(const v8bf*)(bcol + kb);
                ((v8bf*)&bfrag)[1] = *(const v8bf*)(bcol + kb + 16);
                acc = __builtin_amdgcn_wmma_f32_16x16x32_bf16(
                    false, afrag[kt], false, bfrag, (short)0, acc, false, false);
            }
            const float bb = sB1[n];
            const float ww = sW2[n];
            #pragma unroll
            for (int r = 0; r < 8; ++r) {
                float hdn = acc[r] + bb;
                hdn = hdn > 0.0f ? hdn : 0.0f;
                s[r] += hdn * ww;                       // layer-2 partial dot
            }
        }

        // ---- Reduce over 16 output-column lanes, add b2, store ----
        #pragma unroll
        for (int r = 0; r < 8; ++r) {
            float tv = s[r];
            tv += __shfl_xor(tv, 1, 32);
            tv += __shfl_xor(tv, 2, 32);
            tv += __shfl_xor(tv, 4, 32);
            tv += __shfl_xor(tv, 8, 32);                // sum within each 16-lane half
            long long edge = ebase + hi * 8 + r;        // lanes 0-15 -> M=r, 16-31 -> M=8+r
            if ((lane & 15) == 0 && edge < E) {
                out[edge] = tv + b2v;
            }
        }
    }
}

extern "C" void kernel_launch(void* const* d_in, const int* in_sizes, int n_in,
                              void* d_out, int out_size, void* d_ws, size_t ws_size,
                              hipStream_t stream) {
    const float*     h    = (const float*)d_in[0];
    const long long* eidx = (const long long*)d_in[1];   // int64 [2, E]
    const float*     W1   = (const float*)d_in[2];
    const float*     b1   = (const float*)d_in[3];
    const float*     W2   = (const float*)d_in[4];
    const float*     b2   = (const float*)d_in[5];
    float*           out  = (float*)d_out;

    const long long E = (long long)(in_sizes[1] / 2);
    const long long edgesPerBlock = EDGES_PER_BLOCK * TILES_PER_BLOCK;
    const int blocks = (int)((E + edgesPerBlock - 1) / edgesPerBlock);

    link_decoder_kernel<<<blocks, THREADS, SMEM_SIZE, stream>>>(
        h, eidx, W1, b1, W2, b2, out, E);
}
